// Updating_W_Layer_32074815766810
// MI455X (gfx1250) — compile-verified, matrix operand order checked by reading, not run on hardware
//
#include <hip/hip_runtime.h>
#include <hip/hip_bf16.h>

#define ROWS 4096
#define RANK 64
#define COLS 512
#define LAMBDA_TR 0.01f
#define ALPHA 0.1f
#define K1S 1.0f
#define TOLV 1e-8f
#define MAX_ITER 20

typedef __bf16 bf16;
typedef bf16 v16bf  __attribute__((ext_vector_type(16)));
typedef float v8f   __attribute__((ext_vector_type(8)));
typedef unsigned int v4u __attribute__((ext_vector_type(4)));
typedef unsigned int v8u __attribute__((ext_vector_type(8)));

static __device__ __forceinline__ v8f wmma_bf16(v16bf a, v16bf b, v8f c) {
  return __builtin_amdgcn_wmma_f32_16x16x32_bf16(false, a, false, b, (short)0, c, false, false);
}

// CDNA5 wave32 A/B fragment layout: element j <-> K = kbase + (j/8)*16 + half*8 + (j%8)
// => per lane the fragment is two contiguous 16-byte loads. Assemble as dwords so no
// sub-dword moves are emitted; bitcast to the bf16 vector for the WMMA builtin.
static __device__ __forceinline__ v16bf frag_u(const bf16* __restrict__ row, int kbase, int half) {
  v4u lo = *(const v4u*)(row + kbase + half * 8);
  v4u hi = *(const v4u*)(row + kbase + 16 + half * 8);
  v8u u;
#pragma unroll
  for (int j = 0; j < 4; ++j) { u[j] = lo[j]; u[4 + j] = hi[j]; }
  return __builtin_bit_cast(v16bf, u);
}

// Masked variant: binary Omega mask applied as packed bitwise AND (0xFFFF per live slot).
static __device__ __forceinline__ v16bf frag_u_masked(const bf16* __restrict__ row, int kbase,
                                                      int half, v8u m) {
  v4u lo = *(const v4u*)(row + kbase + half * 8);
  v4u hi = *(const v4u*)(row + kbase + 16 + half * 8);
  v8u u;
#pragma unroll
  for (int j = 0; j < 4; ++j) { u[j] = lo[j] & m[j]; u[4 + j] = hi[j] & m[4 + j]; }
  return __builtin_bit_cast(v16bf, u);
}

static __device__ __forceinline__ unsigned maskword(int a, int b) {
  return (a ? 0x0000FFFFu : 0u) | (b ? 0xFFFF0000u : 0u);
}

static __device__ __forceinline__ v8f vzero8() {
  v8f z;
#pragma unroll
  for (int j = 0; j < 8; ++j) z[j] = 0.0f;
  return z;
}

// ---------------- K0: convert / init ----------------
__global__ __launch_bounds__(256) void k_init(
    const float* __restrict__ L, const float* __restrict__ Wpre,
    bf16* __restrict__ Lbf, float* __restrict__ Wa, bf16* __restrict__ WbTa,
    int* __restrict__ doneArr, float* __restrict__ slots) {
  size_t tid = (size_t)blockIdx.x * blockDim.x + threadIdx.x;
  size_t nth = (size_t)gridDim.x * blockDim.x;
  for (size_t idx = tid; idx < (size_t)ROWS * ROWS; idx += nth)
    Lbf[idx] = (bf16)L[idx];
  for (size_t idx = tid; idx < (size_t)ROWS * RANK; idx += nth) {
    float v = Wpre[idx];
    Wa[idx] = v;
    size_t i = idx / RANK, s = idx % RANK;
    WbTa[s * ROWS + i] = (bf16)v;
  }
  if (tid < 21) doneArr[tid] = 0;
  if (tid < 20) slots[tid] = 0.0f;
}

// ---------------- K1: L_W[i] = (H .* Om_i) @ H^T + alpha I, plus V_W ----------------
__global__ __launch_bounds__(128) void k_gram(
    const int* __restrict__ Omega, const float* __restrict__ Hf,
    const float* __restrict__ Z,
    float* __restrict__ LW, float* __restrict__ VW) {
  __shared__ bf16 Hs[RANK * COLS];  // 64KB: shared B operand for every row
  for (int idx = threadIdx.x; idx < RANK * COLS; idx += blockDim.x)
    Hs[idx] = (bf16)Hf[idx];
  __syncthreads();

  const int lane = threadIdx.x & 31;
  const int wave = threadIdx.x >> 5;
  const int i = blockIdx.x * 4 + wave;   // one row per wave
  const int half = lane >> 4;
  const int r16 = lane & 15;

  v8f acc[4][4];
#pragma unroll
  for (int mi = 0; mi < 4; ++mi)
#pragma unroll
    for (int ni = 0; ni < 4; ++ni) acc[mi][ni] = vzero8();

  const int* om = Omega + (size_t)i * COLS;
  for (int k0 = 0; k0 < COLS; k0 += 32) {
    // 16 mask ints covering this lane's K-slots -> 8 packed 32-bit mask words
    int4 o0 = *(const int4*)(om + k0 + half * 8);
    int4 o1 = *(const int4*)(om + k0 + half * 8 + 4);
    int4 o2 = *(const int4*)(om + k0 + 16 + half * 8);
    int4 o3 = *(const int4*)(om + k0 + 16 + half * 8 + 4);
    v8u m;
    m[0] = maskword(o0.x, o0.y); m[1] = maskword(o0.z, o0.w);
    m[2] = maskword(o1.x, o1.y); m[3] = maskword(o1.z, o1.w);
    m[4] = maskword(o2.x, o2.y); m[5] = maskword(o2.z, o2.w);
    m[6] = maskword(o3.x, o3.y); m[7] = maskword(o3.z, o3.w);

    v16bf afr[4], bfr[4];
#pragma unroll
    for (int mi = 0; mi < 4; ++mi)
      afr[mi] = frag_u_masked(Hs + (size_t)(mi * 16 + r16) * COLS, k0, half, m);
#pragma unroll
    for (int ni = 0; ni < 4; ++ni)
      bfr[ni] = frag_u(Hs + (size_t)(ni * 16 + r16) * COLS, k0, half);
#pragma unroll
    for (int mi = 0; mi < 4; ++mi)
#pragma unroll
      for (int ni = 0; ni < 4; ++ni)
        acc[mi][ni] = wmma_bf16(afr[mi], bfr[ni], acc[mi][ni]);
  }

  // store L_W[i] (+ alpha on the diagonal)
  float* LWi = LW + (size_t)i * RANK * RANK;
#pragma unroll
  for (int mi = 0; mi < 4; ++mi)
#pragma unroll
    for (int ni = 0; ni < 4; ++ni)
#pragma unroll
      for (int v = 0; v < 8; ++v) {
        int row = mi * 16 + v + 8 * half;
        int col = ni * 16 + r16;
        float val = acc[mi][ni][v] + (row == col ? ALPHA : 0.0f);
        LWi[row * RANK + col] = val;
      }

  // V_W[i][r] = sum_c Om[i,c]*Z[i,c]*H[r,c]  (f32, lane handles r=lane and r+32)
  const float* zr = Z + (size_t)i * COLS;
  const float* h0 = Hf + (size_t)lane * COLS;
  const float* h1 = Hf + (size_t)(lane + 32) * COLS;
  float vw0 = 0.0f, vw1 = 0.0f;
  for (int c = 0; c < COLS; ++c) {
    float mz = om[c] ? zr[c] : 0.0f;
    vw0 = fmaf(mz, h0[c], vw0);
    vw1 = fmaf(mz, h1[c], vw1);
  }
  VW[(size_t)i * RANK + lane] = vw0;
  VW[(size_t)i * RANK + lane + 32] = vw1;
}

// ---------------- K2: lambda_max via power iteration (SPD + alpha*I) ----------------
__global__ __launch_bounds__(256) void k_lambda(const float* __restrict__ LW,
                                                float* __restrict__ lam) {
  const int lane = threadIdx.x & 31;
  const int wave = threadIdx.x >> 5;
  const int i = blockIdx.x * 8 + wave;
  const float* M = LW + (size_t)i * RANK * RANK;
  float x0 = 1.0f + 0.013f * lane;
  float x1 = 1.0f - 0.007f * lane;
  float nrm = ALPHA;
  for (int it = 0; it < 24; ++it) {
    float y0 = 0.0f, y1 = 0.0f;
    for (int r = 0; r < RANK; ++r) {
      float xr = (r < 32) ? __shfl(x0, r, 32) : __shfl(x1, r - 32, 32);
      y0 = fmaf(xr, M[r * RANK + lane], y0);
      y1 = fmaf(xr, M[r * RANK + lane + 32], y1);
    }
    float s = y0 * y0 + y1 * y1;
#pragma unroll
    for (int off = 16; off; off >>= 1) s += __shfl_xor(s, off, 32);
    nrm = sqrtf(s);                 // ||M x||, x unit -> lambda_max
    float inv = 1.0f / nrm;
    x0 = y0 * inv; x1 = y1 * inv;
  }
  if (lane == 0) lam[i] = nrm;
}

// ---------------- K3: one fixed-point step (MODE 0) or final update (MODE 1) -----------
template <int MODE>
__global__ __launch_bounds__(128) void k_step(
    const bf16* __restrict__ Lbf, const bf16* __restrict__ WbT,
    const float* __restrict__ LW, const float* __restrict__ VW,
    const float* __restrict__ lam,
    const float* __restrict__ Win, float* __restrict__ Wout,
    bf16* __restrict__ WbTout,
    const int* __restrict__ doneArr, float* __restrict__ slots, int t) {
  __shared__ float Ts[4][16][64];   // T = L @ W tile per wave
  const int lane = threadIdx.x & 31;
  const int wave = threadIdx.x >> 5;
  const int w = blockIdx.x * 4 + wave;
  const int i0 = w * 16;
  const int half = lane >> 4;
  const int r16 = lane & 15;

  // ---- GEMM: T[16,64] = L[i0:i0+16, :] @ W  (bf16 WMMA, K = 4096) ----
  v8f acc[4];
#pragma unroll
  for (int ni = 0; ni < 4; ++ni) acc[ni] = vzero8();
  const bf16* arow = Lbf + (size_t)(i0 + r16) * ROWS;
  for (int k0 = 0; k0 < ROWS; k0 += 32) {
    __builtin_prefetch(arow + k0 + 256, 0, 1);
    v16bf a = frag_u(arow, k0, half);
#pragma unroll
    for (int ni = 0; ni < 4; ++ni) {
      v16bf b = frag_u(WbT + (size_t)(ni * 16 + r16) * ROWS, k0, half);
      acc[ni] = wmma_bf16(a, b, acc[ni]);
    }
  }
#pragma unroll
  for (int ni = 0; ni < 4; ++ni)
#pragma unroll
    for (int v = 0; v < 8; ++v)
      Ts[wave][v + 8 * half][ni * 16 + r16] = acc[ni][v];

  // ---- per-row update: W_new = relu(W + (V_W - W@L_W - 0.01*T)/lambda) ----
  const int done = (MODE == 0) ? doneArr[t] : 0;
  float ss = 0.0f;
  for (int rr = 0; rr < 16; ++rr) {
    const int i = i0 + rr;
    const float wv0 = Win[(size_t)i * RANK + lane];
    const float wv1 = Win[(size_t)i * RANK + lane + 32];
    const float* M = LW + (size_t)i * RANK * RANK;
    float l0 = 0.0f, l1 = 0.0f;
    for (int r = 0; r < RANK; ++r) {
      float wr = (r < 32) ? __shfl(wv0, r, 32) : __shfl(wv1, r - 32, 32);
      l0 = fmaf(wr, M[r * RANK + lane], l0);
      l1 = fmaf(wr, M[r * RANK + lane + 32], l1);
    }
    const float inv = 1.0f / lam[i];
    const float t0 = Ts[wave][rr][lane];
    const float t1 = Ts[wave][rr][lane + 32];
    const float v0 = VW[(size_t)i * RANK + lane];
    const float v1 = VW[(size_t)i * RANK + lane + 32];
    float n0 = fmaxf(0.0f, wv0 + (v0 - l0 - LAMBDA_TR * t0) * inv);
    float n1 = fmaxf(0.0f, wv1 + (v1 - l1 - LAMBDA_TR * t1) * inv);
    if (MODE == 1) {
      Wout[(size_t)i * RANK + lane]      = K1S * n0;
      Wout[(size_t)i * RANK + lane + 32] = K1S * n1;
    } else {
      float d0 = n0 - wv0, d1 = n1 - wv1;
      ss += d0 * d0 + d1 * d1;
      float o0 = done ? wv0 : n0;
      float o1 = done ? wv1 : n1;
      Wout[(size_t)i * RANK + lane]      = o0;
      Wout[(size_t)i * RANK + lane + 32] = o1;
      WbTout[(size_t)lane * ROWS + i]        = (bf16)o0;
      WbTout[(size_t)(lane + 32) * ROWS + i] = (bf16)o1;
    }
  }
  if (MODE == 0) {
#pragma unroll
    for (int off = 16; off; off >>= 1) ss += __shfl_xor(ss, off, 32);
    if (lane == 0) atomicAdd(slots + t, ss);
  }
}

// ---------------- K4: convergence flag update ----------------
__global__ void k_flag(const float* __restrict__ slots, int* __restrict__ doneArr, int t) {
  float tc = slots[t] * (1.0f / (float)(RANK * ROWS));
  doneArr[t + 1] = doneArr[t] | (tc <= TOLV ? 1 : 0);
}

extern "C" void kernel_launch(void* const* d_in, const int* in_sizes, int n_in,
                              void* d_out, int out_size, void* d_ws, size_t ws_size,
                              hipStream_t stream) {
  const int*   Omega = (const int*)d_in[0];
  const float* Wpre  = (const float*)d_in[1];
  const float* Hf    = (const float*)d_in[2];
  const float* L     = (const float*)d_in[3];
  const float* Z     = (const float*)d_in[4];
  float* out = (float*)d_out;
  (void)in_sizes; (void)n_in; (void)out_size; (void)ws_size;

  char* ws = (char*)d_ws;
  size_t off = 0;
  auto carve = [&](size_t bytes) -> char* {
    char* p = ws + off;
    off += (bytes + 255) & ~(size_t)255;
    return p;
  };
  bf16*  Lbf   = (bf16*)carve((size_t)ROWS * ROWS * sizeof(bf16));
  float* LW    = (float*)carve((size_t)ROWS * RANK * RANK * sizeof(float));
  float* VW    = (float*)carve((size_t)ROWS * RANK * sizeof(float));
  float* lamv  = (float*)carve((size_t)ROWS * sizeof(float));
  float* Wa    = (float*)carve((size_t)ROWS * RANK * sizeof(float));
  float* Wb    = (float*)carve((size_t)ROWS * RANK * sizeof(float));
  bf16*  WbTa  = (bf16*)carve((size_t)RANK * ROWS * sizeof(bf16));
  bf16*  WbTb  = (bf16*)carve((size_t)RANK * ROWS * sizeof(bf16));
  int*   doneA = (int*)carve(32 * sizeof(int));
  float* slots = (float*)carve(32 * sizeof(float));

  k_init<<<1024, 256, 0, stream>>>(L, Wpre, Lbf, Wa, WbTa, doneA, slots);
  k_gram<<<ROWS / 4, 128, 0, stream>>>(Omega, Hf, Z, LW, VW);
  k_lambda<<<ROWS / 8, 256, 0, stream>>>(LW, lamv);

  float* Win = Wa; float* Wout = Wb;
  bf16* Bin = WbTa; bf16* Bout = WbTb;
  for (int t = 0; t < MAX_ITER - 1; ++t) {
    k_step<0><<<ROWS / 64, 128, 0, stream>>>(Lbf, Bin, LW, VW, lamv,
                                             Win, Wout, Bout, doneA, slots, t);
    k_flag<<<1, 1, 0, stream>>>(slots, doneA, t);
    float* tw = Win; Win = Wout; Wout = tw;
    bf16* tb = Bin; Bin = Bout; Bout = tb;
  }
  // final post-loop update (unconditional, scaled by K1) -> d_out
  k_step<1><<<ROWS / 64, 128, 0, stream>>>(Lbf, Bin, LW, VW, lamv,
                                           Win, out, nullptr, doneA, slots, 0);
}
